// Model_51196010168686
// MI455X (gfx1250) — compile-verified, matrix operand order checked by reading, not run on hardware
//
#include <hip/hip_runtime.h>

typedef __attribute__((ext_vector_type(16))) _Float16 v16h;
typedef __attribute__((ext_vector_type(8)))  _Float16 v8h;
typedef __attribute__((ext_vector_type(8)))  float    v8f;
typedef __attribute__((ext_vector_type(4)))  unsigned int u32x4;
typedef __attribute__((ext_vector_type(8)))  int          i32x8;
typedef __attribute__((ext_vector_type(4)))  int          i32x4;

#define BM 128
#define BN 128
#define BK 32
#define LDT 40   // LDS row stride in halves: 64B row + 16B TDM pad = 80B

// ---------------- prep kernels ----------------

__global__ __launch_bounds__(256) void cvt_x_f16(const float* __restrict__ x,
                                                 _Float16* __restrict__ xh,
                                                 long n) {
    long i = ((long)blockIdx.x * 256 + threadIdx.x) * 8;
    if (i + 8 > n) return;
    const float4* p = (const float4*)(x + i);
    float4 a = p[0], b = p[1];
    v8h o;
    o[0] = (_Float16)a.x; o[1] = (_Float16)a.y;
    o[2] = (_Float16)a.z; o[3] = (_Float16)a.w;
    o[4] = (_Float16)b.x; o[5] = (_Float16)b.y;
    o[6] = (_Float16)b.z; o[7] = (_Float16)b.w;
    *(v8h*)(xh + i) = o;
}

// per-group(8) symmetric int4 quant-dequant along in_features, fp32 -> f16,
// layout preserved as [out][in] (== B in N-major form for the GEMM)
__global__ __launch_bounds__(256) void dequant_w_f16(const float* __restrict__ w,
                                                     _Float16* __restrict__ wh,
                                                     long ngroups) {
    long g = (long)blockIdx.x * 256 + threadIdx.x;
    if (g >= ngroups) return;
    long base = g * 8;
    const float4* p = (const float4*)(w + base);
    float4 a = p[0], b = p[1];
    float v[8] = {a.x, a.y, a.z, a.w, b.x, b.y, b.z, b.w};
    float m = 0.0f;
#pragma unroll
    for (int i = 0; i < 8; ++i) m = fmaxf(m, fabsf(v[i]));
    float scale = fmaxf(m * (1.0f / 7.0f), 1e-8f);
    float inv = 1.0f / scale;
    v8h o;
#pragma unroll
    for (int i = 0; i < 8; ++i) {
        float q = rintf(v[i] * inv);
        q = fminf(7.0f, fmaxf(-7.0f, q));
        o[i] = (_Float16)(q * scale);
    }
    *(v8h*)(wh + base) = o;
}

// ---------------- TDM descriptor helpers ----------------

union G0u { unsigned long long q[2]; u32x4 v; };
union G1u { unsigned long long q[4]; i32x8 v; };

// D# group1 for a BMxBK f16 tile of a row-major [rows][K] tensor:
//   data_size=1 (2B), pad_enable, pad_interval=3 (16 DWORDs = one 64B row),
//   pad_amount=3 (4 DWORDs = 16B) -> LDS row stride 80B (= LDT halves),
//   tensor_dim0=K, tensor_dim1=big, tile_dim0=BK, tile_dim1=BM,
//   tensor_dim0_stride=K
__device__ __forceinline__ i32x8 tdm_make_g1(int K) {
    G1u g1;
    g1.q[0] = 0x06D10000ULL | ((unsigned long long)(K & 0xFFFF) << 48);
    g1.q[1] = (unsigned long long)((unsigned)K >> 16)
            | (((unsigned long long)(1u << 20)) << 16)      // tensor_dim1
            | ((unsigned long long)BK << 48);               // tile_dim0
    g1.q[2] = (unsigned long long)BM                        // tile_dim1
            | (((unsigned long long)(unsigned)K) << 32);    // dim0_stride lo
    g1.q[3] = 0ULL;                                         // 2D: dim1_stride unused
    return g1.v;
}

__device__ __forceinline__ void tdm_load_tile(unsigned int lds_off,
                                              const _Float16* gptr,
                                              i32x8 g1) {
    G0u g0;
    g0.q[0] = 1ULL | ((unsigned long long)lds_off << 32);   // count=1 | lds_addr
    g0.q[1] = ((unsigned long long)(uintptr_t)gptr & 0x01FFFFFFFFFFFFFFULL)
            | (2ULL << 62);                                 // global_addr | type=2
    i32x4 z4 = {};
#if __clang_major__ >= 23
    i32x8 z8 = {};
    __builtin_amdgcn_tensor_load_to_lds(g0.v, g1, z4, z4, z8, 0);
#else
    __builtin_amdgcn_tensor_load_to_lds(g0.v, g1, z4, z4, 0);
#endif
}

__device__ __forceinline__ unsigned int lds_off32(const void* p) {
    return (unsigned int)(uintptr_t)p;   // LDS aperture: offset = addr[31:0]
}

// ---------------- WMMA GEMM ----------------
// C[M][N] = A[M][K] * B[N][K]^T + bias, f16 inputs, f32 accum.
// Tiles staged by the Tensor Data Mover (TENSORcnt), double-buffered in LDS.
__global__ __launch_bounds__(256) void gemm_f16_wmma(
    const _Float16* __restrict__ A,   // [M][K]
    const _Float16* __restrict__ B,   // [N][K]
    const float*    __restrict__ bias,// [N]
    float*          __restrict__ C,   // [M][N]
    int M, int N, int K)
{
    __shared__ _Float16 sA[2][BM * LDT];
    __shared__ _Float16 sB[2][BN * LDT];

    const int tid  = threadIdx.x;
    const int lane = tid & 31;
    const int wave = tid >> 5;
    const int wm   = wave >> 2;     // 0..1  (64-row slab)
    const int wn   = wave & 3;      // 0..3  (32-col slab)
    const int r16  = lane & 15;
    const int khalf = lane >> 4;    // 0/1 -> K offset 0/8 within fragment

    const int m0 = blockIdx.y * BM;
    const int n0 = blockIdx.x * BN;

    const _Float16* aPtr = A + (long)m0 * K;
    const _Float16* bPtr = B + (long)n0 * K;

    const i32x8 g1 = tdm_make_g1(K);

    v8f zero = {};
    v8f acc[4][2];
#pragma unroll
    for (int mt = 0; mt < 4; ++mt)
#pragma unroll
        for (int nt = 0; nt < 2; ++nt) acc[mt][nt] = zero;

    // stage tile 0 via TDM (one wave issues; EXEC-independent DMA)
    if (wave == 0) {
        tdm_load_tile(lds_off32(&sA[0][0]), aPtr, g1);
        tdm_load_tile(lds_off32(&sB[0][0]), bPtr, g1);
        __builtin_amdgcn_s_wait_tensorcnt(0);
    }
    __syncthreads();

    const int nk = K / BK;
    for (int kt = 0; kt < nk; ++kt) {
        const int buf = kt & 1;

        // issue DMA for tile kt+1 into the other buffer; its last readers
        // finished before the barrier that ended iteration kt-1.
        if (wave == 0 && kt + 1 < nk) {
            tdm_load_tile(lds_off32(&sA[buf ^ 1][0]), aPtr + (kt + 1) * BK, g1);
            tdm_load_tile(lds_off32(&sB[buf ^ 1][0]), bPtr + (kt + 1) * BK, g1);
        }

        // fragment loads from LDS (ISA 16-bit A-matrix 16x32 layout:
        // lane%16 = row, lane/16 selects K {0..7,16..23} vs {8..15,24..31})
        const _Float16* sa = sA[buf];
        const _Float16* sb = sB[buf];
        v16h aF[4], bF[2];
#pragma unroll
        for (int mt = 0; mt < 4; ++mt) {
            int row = (wm * 64 + mt * 16 + r16) * LDT + khalf * 8;
            v8h lo = *(const v8h*)(sa + row);
            v8h hi = *(const v8h*)(sa + row + 16);
            aF[mt] = __builtin_shufflevector(lo, hi, 0,1,2,3,4,5,6,7,8,9,10,11,12,13,14,15);
        }
#pragma unroll
        for (int nt = 0; nt < 2; ++nt) {
            int row = (wn * 32 + nt * 16 + r16) * LDT + khalf * 8;
            v8h lo = *(const v8h*)(sb + row);
            v8h hi = *(const v8h*)(sb + row + 16);
            bF[nt] = __builtin_shufflevector(lo, hi, 0,1,2,3,4,5,6,7,8,9,10,11,12,13,14,15);
        }

#pragma unroll
        for (int mt = 0; mt < 4; ++mt)
#pragma unroll
            for (int nt = 0; nt < 2; ++nt)
                acc[mt][nt] = __builtin_amdgcn_wmma_f32_16x16x32_f16(
                    false, aF[mt], false, bF[nt],
                    (short)0, acc[mt][nt], false, false);

        // wave 0 confirms the DMA for tile kt+1 landed, then everyone syncs
        if (wave == 0) __builtin_amdgcn_s_wait_tensorcnt(0);
        __syncthreads();
    }

    // epilogue: C/D layout -> lane%16 = col, VGPR j + (lane/16)*8 = row; fuse bias
#pragma unroll
    for (int nt = 0; nt < 2; ++nt) {
        int col = n0 + wn * 32 + nt * 16 + r16;
        float bv = bias[col];
#pragma unroll
        for (int mt = 0; mt < 4; ++mt) {
            int mbase = m0 + wm * 64 + mt * 16 + khalf * 8;
#pragma unroll
            for (int j = 0; j < 8; ++j)
                C[(long)(mbase + j) * N + col] = acc[mt][nt][j] + bv;
        }
    }
}

// ---------------- launcher ----------------

extern "C" void kernel_launch(void* const* d_in, const int* in_sizes, int n_in,
                              void* d_out, int out_size, void* d_ws, size_t ws_size,
                              hipStream_t stream) {
    const float* x    = (const float*)d_in[0];
    const float* w    = (const float*)d_in[1];
    const float* bias = (const float*)d_in[2];
    float* out = (float*)d_out;

    const long dout   = in_sizes[2];                  // 4096
    const long din    = (long)in_sizes[1] / dout;     // 4096
    const long tokens = (long)in_sizes[0] / din;      // 8192
    const int M = (int)tokens, N = (int)dout, K = (int)din;

    _Float16* xh = (_Float16*)d_ws;                   // [M][K] f16 (64 MB)
    _Float16* wh = xh + (size_t)M * K;                // [N][K] f16 (32 MB)

    long nx = (long)M * K;
    cvt_x_f16<<<(int)(nx / (256 * 8)), 256, 0, stream>>>(x, xh, nx);

    long ng = (long)N * K / 8;
    dequant_w_f16<<<(int)(ng / 256), 256, 0, stream>>>(w, wh, ng);

    dim3 grid(N / BN, M / BM);
    gemm_f16_wmma<<<grid, 256, 0, stream>>>(xh, wh, bias, out, M, N, K);
}
